// my_gat_22832046146094
// MI455X (gfx1250) — matrix operand order repaired and checked by reading, not compiled, not run on hardware
//
#include <hip/hip_runtime.h>
#include <math.h>

typedef __attribute__((ext_vector_type(16))) _Float16 v16h;
typedef __attribute__((ext_vector_type(8)))  float    v8f;
typedef __attribute__((ext_vector_type(4)))  float    v4f;

#define GG 48      // B*P graphs
#define NN 1024    // nodes
#define CC 64      // channels
#define LOG2E 1.44269504088896340736f

static __device__ __forceinline__ v8f wmma_f16(v16h a, v16h b, v8f c) {
  // D(16x16 f32) = A(16x32 f16) * B(32x16 f16) + C
  return __builtin_amdgcn_wmma_f32_16x16x32_f16(false, a, false, b, (short)0, c,
                                                false, false);
}

static __device__ __forceinline__ float lrelu01(float z) {
  // leaky_relu slope 0.01: for z<0, 0.01z > z; for z>=0, z >= 0.01z
  return fmaxf(z, 0.01f * z);
}

static __device__ __forceinline__ float exp2_fast(float x) {
#if __has_builtin(__builtin_amdgcn_exp2f)
  return __builtin_amdgcn_exp2f(x);   // bare v_exp_f32 (2^x)
#else
  return exp2f(x);
#endif
}

static __device__ __forceinline__ float rcp_fast(float x) {
#if __has_builtin(__builtin_amdgcn_rcpf)
  return __builtin_amdgcn_rcpf(x);    // v_rcp_f32
#else
  return 1.0f / x;
#endif
}

// ---------------------------------------------------------------------------
// Kernel 0: fold a1/a2 through W_emb (s = x_emb @ a == x @ (a @ W_emb)),
// and convert W_lin to f16 (row-major [o][c], which is exactly the WMMA
// B-operand source layout for the epilogue GEMM).
// ---------------------------------------------------------------------------
__global__ void prep_params(const float* __restrict__ W_emb,
                            const float* __restrict__ a1,
                            const float* __restrict__ a2,
                            const float* __restrict__ W_lin,
                            float* __restrict__ u1, float* __restrict__ u2,
                            _Float16* __restrict__ WlH) {
  const int c = threadIdx.x;  // 0..63
  float su1 = 0.f, su2 = 0.f;
  #pragma unroll 4
  for (int h = 0; h < CC; ++h) {
    const float w = W_emb[h * CC + c];
    su1 += a1[h] * w;
    su2 += a2[h] * w;
  }
  u1[c] = su1;
  u2[c] = su2;
  #pragma unroll 4
  for (int k = 0; k < CC; ++k) {
    const int idx = k * CC + c;
    WlH[idx] = (_Float16)W_lin[idx];
  }
}

// ---------------------------------------------------------------------------
// Kernel 1: per node -> s1 = x.u1, s2 = x.u2 (PRE-SCALED by log2(e): lrelu is
// positively homogeneous, so the whole score pipeline can run in the log2
// domain and the softmax exp becomes a bare v_exp_f32), plus transpose x into
// f16 xT[g][c][n] so WMMA B-operands are contiguous 32B loads along K.
// ---------------------------------------------------------------------------
__global__ void prep_x(const float* __restrict__ x,
                       const float* __restrict__ u1,
                       const float* __restrict__ u2,
                       float* __restrict__ s1, float* __restrict__ s2,
                       _Float16* __restrict__ xT) {
  __shared__ float su1[CC], su2[CC];
  if (threadIdx.x < CC) {
    su1[threadIdx.x] = u1[threadIdx.x];
    su2[threadIdx.x] = u2[threadIdx.x];
  }
  __syncthreads();
  const int gi = blockIdx.x * blockDim.x + threadIdx.x;  // global node id
  const int g = gi >> 10;
  const int i = gi & (NN - 1);
  const float* xr = x + (size_t)gi * CC;
  float d1 = 0.f, d2 = 0.f;
  #pragma unroll
  for (int c = 0; c < CC; ++c) {
    const float v = xr[c];
    d1 += v * su1[c];
    d2 += v * su2[c];
  }
  s1[gi] = d1 * LOG2E;
  s2[gi] = d2 * LOG2E;
  _Float16* xtg = xT + (size_t)g * CC * NN + i;
  #pragma unroll
  for (int c = 0; c < CC; ++c) xtg[(size_t)c * NN] = (_Float16)xr[c];
}

// ---------------------------------------------------------------------------
// Kernel 2: fused masked-softmax + aggregation + output projection.
// One wave owns a 16-row tile of one graph.
//   Sweep 1: copy A row -> d_out (required output, NT stores) while taking
//            the masked max of s2 (lrelu monotonic => rowmax = lrelu(s1+max)).
//   Sweep 2: (L2-hot re-read of A) build exp2 tiles branchlessly, directly in
//            the WMMA A-operand register layout, accumulate the softmax
//            denominator as a side product; 4 WMMAs per 32-col chunk.
//   Tail:    rescale by 1/l, stage agg f16 in LDS, 8 WMMAs vs W_lin f16,
//            sigmoid, store.
// ---------------------------------------------------------------------------
__global__ __launch_bounds__(128) void gat_main(
    const float* __restrict__ A,
    const float* __restrict__ s1buf,
    const float* __restrict__ s2buf,
    const _Float16* __restrict__ xT,
    const _Float16* __restrict__ WlH,
    float* __restrict__ out,
    float* __restrict__ outA) {
  __shared__ __align__(16) float s2sh[NN];
  __shared__ float invlsh[4][16];
  __shared__ __align__(16) _Float16 aggsh[4][16 * CC];

  const int tid = threadIdx.x;
  const int lane = tid & 31;
  const int wv = tid >> 5;        // wave in block (4 waves)
  const int col = lane & 15;      // row-in-tile for A-operand, N-col for B/C/D
  const int hf = lane >> 4;       // K-half select
  const int kb = 8 * hf;
  const int g = blockIdx.x >> 4;
  const int i0 = ((blockIdx.x & 15) * 4 + wv) * 16;  // first row of tile

  // stage s2 (log2-scaled) for this graph in LDS
  {
    const v4f* src = (const v4f*)(s2buf + (size_t)g * NN);
    v4f* dst = (v4f*)s2sh;
    for (int t = tid; t < NN / 4; t += 128) dst[t] = src[t];
  }
  __syncthreads();

  const size_t rowOff = ((size_t)g * NN + (size_t)(i0 + col)) * NN;
  const float* Arow = A + rowOff;
  float* Acpy = outA + rowOff;

  // -------- Sweep 1: masked max of s2 + A copy-out (NT stores) --------
  float s2m = -3.0e38f;
  {
    const int jbase = hf * 512;  // lane pair (m, m+16) splits the row
    #pragma unroll 4
    for (int jj = 0; jj < 512; jj += 4) {
      const v4f av = *(const v4f*)(Arow + jbase + jj);
      __builtin_nontemporal_store(av, (v4f*)(Acpy + jbase + jj));
      const v4f sv = *(const v4f*)(s2sh + jbase + jj);
      s2m = fmaxf(s2m, (av[0] != 0.f) ? sv[0] : -3.0e38f);
      s2m = fmaxf(s2m, (av[1] != 0.f) ? sv[1] : -3.0e38f);
      s2m = fmaxf(s2m, (av[2] != 0.f) ? sv[2] : -3.0e38f);
      s2m = fmaxf(s2m, (av[3] != 0.f) ? sv[3] : -3.0e38f);
    }
  }
  s2m = fmaxf(s2m, __shfl_xor(s2m, 16, 32));
  const float s1v = s1buf[(size_t)g * NN + i0 + col];
  const float mi = lrelu01(s1v + s2m);  // log2-domain row max (lrelu monotonic)

  // -------- Sweep 2: exp2 tiles + WMMA aggregation over x --------
  v8f acc0 = {}, acc1 = {}, acc2 = {}, acc3 = {};
  float lsum = 0.f;
  const _Float16* xTg = xT + (size_t)g * CC * NN;
  for (int jc = 0; jc < NN; jc += 32) {
    // one contiguous 16-element K run per lane: k = 16*hf + e (same order as B)
    const float* Ar1 = Arow + jc + 16 * hf;
    const float* Ar2 = Arow + jc + 16 * hf + 8;
    const float* Sp1 = s2sh + jc + 16 * hf;
    const float* Sp2 = s2sh + jc + 16 * hf + 8;
    const v4f m0 = *(const v4f*)(Ar1);
    const v4f m1 = *(const v4f*)(Ar1 + 4);
    const v4f m2 = *(const v4f*)(Ar2);
    const v4f m3 = *(const v4f*)(Ar2 + 4);
    const v4f t0 = *(const v4f*)(Sp1);
    const v4f t1 = *(const v4f*)(Sp1 + 4);
    const v4f t2 = *(const v4f*)(Sp2);
    const v4f t3 = *(const v4f*)(Sp2 + 4);
    const float ms[16] = {m0[0], m0[1], m0[2], m0[3], m1[0], m1[1], m1[2], m1[3],
                          m2[0], m2[1], m2[2], m2[3], m3[0], m3[1], m3[2], m3[3]};
    const float ss[16] = {t0[0], t0[1], t0[2], t0[3], t1[0], t1[1], t1[2], t1[3],
                          t2[0], t2[1], t2[2], t2[3], t3[0], t3[1], t3[2], t3[3]};
    float ps[16];
    v16h pa;
    #pragma unroll
    for (int e = 0; e < 16; ++e) {
      const float z = lrelu01(s1v + ss[e]);          // log2 domain
      // branchless mask: exp2(-1e30) flushes to 0, no exec-mask divergence
      const float zarg = (ms[e] != 0.f) ? (z - mi) : -1.0e30f;
      const float p = exp2_fast(zarg);               // bare v_exp_f32
      ps[e] = p;
      pa[e] = (_Float16)p;
    }
    // tree-reduce denominator (avoid 16-deep serial dependency chain)
    {
      const float l0 = (ps[0] + ps[1]) + (ps[2] + ps[3]);
      const float l1 = (ps[4] + ps[5]) + (ps[6] + ps[7]);
      const float l2 = (ps[8] + ps[9]) + (ps[10] + ps[11]);
      const float l3 = (ps[12] + ps[13]) + (ps[14] + ps[15]);
      lsum += (l0 + l1) + (l2 + l3);
    }
    const _Float16* xb = xTg + (size_t)col * NN + jc + 16 * hf;
    const v16h b0 = *(const v16h*)(xb);
    const v16h b1 = *(const v16h*)(xb + 16 * NN);
    const v16h b2 = *(const v16h*)(xb + 32 * NN);
    const v16h b3 = *(const v16h*)(xb + 48 * NN);
    acc0 = wmma_f16(pa, b0, acc0);
    acc1 = wmma_f16(pa, b1, acc1);
    acc2 = wmma_f16(pa, b2, acc2);
    acc3 = wmma_f16(pa, b3, acc3);
  }

  // softmax denominator: combine lane pair, broadcast 1/l via LDS
  lsum += __shfl_xor(lsum, 16, 32);
  if (hf == 0) invlsh[wv][col] = rcp_fast(lsum);
  __syncthreads();

  // rescale + stage agg tile as f16 in LDS (C/D layout -> row-major)
  #pragma unroll
  for (int r = 0; r < 8; ++r) {
    const int rowm = r + 8 * hf;  // C/D layout: lanes 16-31 hold M=8..15
    const float sc = invlsh[wv][rowm];
    _Float16* ar = &aggsh[wv][rowm * CC + col];
    ar[0]  = (_Float16)(acc0[r] * sc);
    ar[16] = (_Float16)(acc1[r] * sc);
    ar[32] = (_Float16)(acc2[r] * sc);
    ar[48] = (_Float16)(acc3[r] * sc);
  }
  __syncthreads();

  // -------- Epilogue: out = sigmoid(agg @ W_lin^T), M=16 K=64 N=64 --------
  v8f o0 = {}, o1 = {}, o2 = {}, o3 = {};
  #pragma unroll
  for (int kc = 0; kc < CC; kc += 32) {
    const _Float16* ag = &aggsh[wv][col * CC + kc + 16 * hf];
    union { v16h h; uint4 u[2]; } ua;
    ua.u[0] = *(const uint4*)(ag);
    ua.u[1] = *(const uint4*)(ag + 8);
    const _Float16* wb = WlH + (size_t)col * CC + kc + 16 * hf;
    const v16h w0 = *(const v16h*)(wb);
    const v16h w1 = *(const v16h*)(wb + 16 * CC);
    const v16h w2 = *(const v16h*)(wb + 32 * CC);
    const v16h w3 = *(const v16h*)(wb + 48 * CC);
    o0 = wmma_f16(ua.h, w0, o0);
    o1 = wmma_f16(ua.h, w1, o1);
    o2 = wmma_f16(ua.h, w2, o2);
    o3 = wmma_f16(ua.h, w3, o3);
  }
  float* outg = out + ((size_t)g * NN + i0) * CC;
  #pragma unroll
  for (int r = 0; r < 8; ++r) {
    const int rowm = r + 8 * hf;
    float* orow = outg + (size_t)rowm * CC + col;
    // sigmoid(x) = 1/(1+2^(-x*log2e))
    orow[0]  = rcp_fast(1.f + exp2_fast(-o0[r] * LOG2E));
    orow[16] = rcp_fast(1.f + exp2_fast(-o1[r] * LOG2E));
    orow[32] = rcp_fast(1.f + exp2_fast(-o2[r] * LOG2E));
    orow[48] = rcp_fast(1.f + exp2_fast(-o3[r] * LOG2E));
  }
}

// ---------------------------------------------------------------------------
extern "C" void kernel_launch(void* const* d_in, const int* in_sizes, int n_in,
                              void* d_out, int out_size, void* d_ws,
                              size_t ws_size, hipStream_t stream) {
  (void)in_sizes; (void)n_in; (void)out_size; (void)ws_size;
  const float* x     = (const float*)d_in[0];
  const float* A     = (const float*)d_in[1];
  const float* W_emb = (const float*)d_in[2];
  const float* a1    = (const float*)d_in[3];
  const float* a2    = (const float*)d_in[4];
  const float* W_lin = (const float*)d_in[5];

  float* out  = (float*)d_out;
  float* outA = out + (size_t)GG * NN * CC;  // tuple output #2: A passthrough

  // workspace carve-up (~6.7 MB total)
  float*    u1  = (float*)d_ws;                                    // 64 f32
  float*    u2  = u1 + 64;                                         // 64 f32
  _Float16* WlH = (_Float16*)(u2 + 64);                            // 4096 f16
  float*    s1  = (float*)((char*)WlH + 4096 * sizeof(_Float16));  // G*N f32
  float*    s2  = s1 + (size_t)GG * NN;                            // G*N f32
  _Float16* xT  = (_Float16*)(s2 + (size_t)GG * NN);               // G*C*N f16

  prep_params<<<1, 64, 0, stream>>>(W_emb, a1, a2, W_lin, u1, u2, WlH);
  prep_x<<<(GG * NN) / 256, 256, 0, stream>>>(x, u1, u2, s1, s2, xT);
  gat_main<<<GG * (NN / 64), 128, 0, stream>>>(A, s1, s2, xT, WlH, out, outA);
}
